// MultiHeadLatentAttention_70617852281235
// MI455X (gfx1250) — compile-verified, matrix operand order checked by reading, not run on hardware
//
#include <hip/hip_runtime.h>
#include <hip/hip_bf16.h>
#include <stdint.h>

// ---------------- problem constants ----------------
#define B_SZ    2
#define S_LEN   2048
#define HEADS   16
#define DMODEL  2048
#define DLAT    512
#define DH      128
#define DHR     64
#define DQK     192
#define NTOK    (B_SZ * S_LEN)          // 4096 tokens
#define QW      8                       // q-tile waves per attention block

#if __has_builtin(__builtin_amdgcn_global_load_async_to_lds_b128) && \
    __has_builtin(__builtin_amdgcn_s_wait_asynccnt)
#define ASYNC_LDS 1
#endif

typedef __bf16 bf16_t;
typedef __attribute__((ext_vector_type(16))) __bf16 bf16x16;
typedef __attribute__((ext_vector_type(8)))  float  f32x8;

// exact parameter types for the async global->LDS builtin (from the clang
// diagnostic: non-const int4-vector pointers; AS1 source, AS3 destination)
typedef int v4i_ty __attribute__((vector_size(16)));
typedef __attribute__((address_space(1))) v4i_ty as1_v4i;
typedef __attribute__((address_space(3))) v4i_ty as3_v4i;

// ---------------- bf16 helpers (bit-exact RNE) ----------------
__device__ __forceinline__ float bf2f(bf16_t b) {
    union { unsigned short s; bf16_t b; } u; u.b = b;
    return __uint_as_float(((unsigned int)u.s) << 16);
}
__device__ __forceinline__ bf16_t f2bf(float f) {
    unsigned int u = __float_as_uint(f);
    unsigned int r = u + 0x7FFFu + ((u >> 16) & 1u);
    union { unsigned short s; bf16_t b; } v; v.s = (unsigned short)(r >> 16);
    return v.b;
}
__device__ __forceinline__ f32x8 zero8() {
    f32x8 z;
#pragma unroll
    for (int i = 0; i < 8; ++i) z[i] = 0.0f;
    return z;
}

// Load a 16x32 bf16 tile fragment (A-layout == B-layout for X @ W^T):
// lane L -> row (L&15); element e -> k = (e>=8?16:0) + (L>=16?8:0) + (e&7).
__device__ __forceinline__ bf16x16 load_frag(const bf16_t* __restrict__ base,
                                             int ld, int lane) {
    const int r  = lane & 15;
    const int ko = (lane >> 4) << 3;       // 0 or 8
    const bf16_t* p = base + (size_t)r * ld + ko;
    union { uint4 q; bf16_t e[8]; } lo, hi;
    lo.q = *reinterpret_cast<const uint4*>(p);
    hi.q = *reinterpret_cast<const uint4*>(p + 16);
    bf16x16 f;
#pragma unroll
    for (int i = 0; i < 8; ++i) { f[i] = lo.e[i]; f[i + 8] = hi.e[i]; }
    return f;
}

__device__ __forceinline__ f32x8 wmma_bf16(bf16x16 a, bf16x16 b, f32x8 c) {
    return __builtin_amdgcn_wmma_f32_16x16x32_bf16(
        /*neg_a=*/false, a, /*neg_b=*/false, b,
        /*c_mod=*/(short)0, c, /*reuse_a=*/false, /*reuse_b=*/false);
}

// 16-byte global->LDS copy: async TDM-class path when available.
__device__ __forceinline__ void cp16(const bf16_t* g, bf16_t* l) {
#if defined(ASYNC_LDS)
    __builtin_amdgcn_global_load_async_to_lds_b128(
        (as1_v4i*)g, (as3_v4i*)l, 0, 0);
#else
    *reinterpret_cast<uint4*>(l) = *reinterpret_cast<const uint4*>(g);
#endif
}
__device__ __forceinline__ void cp16_drain() {
#if defined(ASYNC_LDS)
    __builtin_amdgcn_s_wait_asynccnt(0);
#endif
}

// ---------------- fp32 -> bf16 convert ----------------
__global__ void cvt_f32_bf16(const float* __restrict__ in,
                             bf16_t* __restrict__ out, long n) {
    long i = (long)blockIdx.x * blockDim.x + threadIdx.x;
    long stride = (long)gridDim.x * blockDim.x;
    for (; i < n; i += stride) out[i] = f2bf(in[i]);
}

// ---------------- generic GEMM: C[MxN] = A[MxK] @ W[NxK]^T ----------------
// One wave per 32x64 strip, 8 accumulators. K unrolled by 2 with ping-pong
// fragment register sets: no cur<-nxt copies, loads overlap WMMA via partial
// loadcnt waits. Requires K % 64 == 0 (true for all call sites).
__global__ __launch_bounds__(32)
void gemm_bf16_wmma(const bf16_t* __restrict__ A,
                    const bf16_t* __restrict__ W,
                    bf16_t* __restrict__ Cb,   // bf16 out (or null)
                    float*  __restrict__ Cf,   // f32 out  (or null)
                    int M, int N, int K) {
    const int lane   = threadIdx.x & 31;
    const int tileN0 = blockIdx.x * 64;
    const int tileM0 = blockIdx.y * 32;

    const bf16_t* Arow[2];
    Arow[0] = A + (size_t)tileM0 * K;
    Arow[1] = A + (size_t)(tileM0 + 16) * K;
    const bf16_t* Wrow[4];
#pragma unroll
    for (int t = 0; t < 4; ++t) Wrow[t] = W + (size_t)(tileN0 + t * 16) * K;

    f32x8 acc[2][4];
#pragma unroll
    for (int m = 0; m < 2; ++m)
#pragma unroll
        for (int t = 0; t < 4; ++t) acc[m][t] = zero8();

    bf16x16 a0[2], b0[4], a1[2], b1[4];
#pragma unroll
    for (int m = 0; m < 2; ++m) a0[m] = load_frag(Arow[m], K, lane);
#pragma unroll
    for (int t = 0; t < 4; ++t) b0[t] = load_frag(Wrow[t], K, lane);

    for (int k0 = 0; k0 < K; k0 += 64) {
        const int kA = k0 + 32;                       // always < K (K%64==0)
#pragma unroll
        for (int m = 0; m < 2; ++m) a1[m] = load_frag(Arow[m] + kA, K, lane);
#pragma unroll
        for (int t = 0; t < 4; ++t) b1[t] = load_frag(Wrow[t] + kA, K, lane);
#pragma unroll
        for (int m = 0; m < 2; ++m)
#pragma unroll
            for (int t = 0; t < 4; ++t)
                acc[m][t] = wmma_bf16(a0[m], b0[t], acc[m][t]);

        const int kB = (k0 + 64 < K) ? (k0 + 64) : 0; // wrap: harmless warm-up
#pragma unroll
        for (int m = 0; m < 2; ++m) a0[m] = load_frag(Arow[m] + kB, K, lane);
#pragma unroll
        for (int t = 0; t < 4; ++t) b0[t] = load_frag(Wrow[t] + kB, K, lane);
#pragma unroll
        for (int m = 0; m < 2; ++m)
#pragma unroll
            for (int t = 0; t < 4; ++t)
                acc[m][t] = wmma_bf16(a1[m], b1[t], acc[m][t]);
    }

    const int col0 = lane & 15;
    const int rowh = (lane >> 4) << 3;
#pragma unroll
    for (int m = 0; m < 2; ++m)
#pragma unroll
        for (int t = 0; t < 4; ++t)
#pragma unroll
            for (int v = 0; v < 8; ++v) {
                size_t row = (size_t)tileM0 + m * 16 + rowh + v;
                size_t col = (size_t)tileN0 + t * 16 + col0;
                if (Cf) Cf[row * N + col] = acc[m][t][v];
                else    Cb[row * N + col] = f2bf(acc[m][t][v]);
            }
}

// ---------------- assemble: RoPE + concat + V transpose ----------------
__global__ void mla_assemble(const bf16_t* __restrict__ Qn,
                             const bf16_t* __restrict__ Qr,
                             const bf16_t* __restrict__ Kn,
                             const bf16_t* __restrict__ Kr,
                             const bf16_t* __restrict__ Vn,
                             bf16_t* __restrict__ Qf,
                             bf16_t* __restrict__ Kf,
                             bf16_t* __restrict__ Vt) {
    int idx = blockIdx.x * blockDim.x + threadIdx.x;   // (b*H + h)*S + s
    if (idx >= B_SZ * HEADS * S_LEN) return;
    const int s = idx % S_LEN;
    const int h = (idx / S_LEN) % HEADS;
    const int b = idx / (S_LEN * HEADS);
    const size_t tok = (size_t)b * S_LEN + s;

    bf16_t* qrow = Qf + (size_t)idx * DQK;
    bf16_t* krow = Kf + (size_t)idx * DQK;
    const bf16_t* qn = Qn + tok * DMODEL + (size_t)h * DH;
    const bf16_t* kn = Kn + tok * DMODEL + (size_t)h * DH;
    const bf16_t* vn = Vn + tok * DMODEL + (size_t)h * DH;
#pragma unroll 4
    for (int i = 0; i < DH; ++i) { qrow[i] = qn[i]; krow[i] = kn[i]; }

    const bf16_t* qr = Qr + tok * (HEADS * DHR) + (size_t)h * DHR;
    const bf16_t* kr = Kr + tok * DHR;          // shared across heads
    const float fs = (float)s;
#pragma unroll 4
    for (int i = 0; i < DHR / 2; ++i) {
        float inv_freq = __powf(10000.0f, -(float)i / (DHR / 2));
        float sn, cs;
        __sincosf(fs * inv_freq, &sn, &cs);
        float x1 = bf2f(qr[2 * i]), x2 = bf2f(qr[2 * i + 1]);
        qrow[DH + 2 * i]     = f2bf(x1 * cs - x2 * sn);
        qrow[DH + 2 * i + 1] = f2bf(x1 * sn + x2 * cs);
        float y1 = bf2f(kr[2 * i]), y2 = bf2f(kr[2 * i + 1]);
        krow[DH + 2 * i]     = f2bf(y1 * cs - y2 * sn);
        krow[DH + 2 * i + 1] = f2bf(y1 * sn + y2 * cs);
    }

    bf16_t* vt = Vt + ((size_t)(b * HEADS + h) * DH) * S_LEN + s;
#pragma unroll 4
    for (int i = 0; i < DH; ++i) vt[(size_t)i * S_LEN] = vn[i];
}

// ---------------- flash attention, block-shared K/V ----------------
// 8 waves/block, each owns a 16-query tile of the same (b,h). Key tiles of 32
// are staged once into double-buffered LDS (async global->LDS when available)
// and consumed by all 8 waves: 8x less K/V read traffic than per-wave streams.
// Fully-masked tiles are exact no-ops in the online softmax (alpha=1, p=0).
__global__ __launch_bounds__(256)
void mla_attention(const bf16_t* __restrict__ Qf,
                   const bf16_t* __restrict__ Kf,
                   const bf16_t* __restrict__ Vt,
                   bf16_t* __restrict__ Att) {
    __shared__ __align__(16) bf16_t Kl[2][32 * DQK];   // 2 x 12 KB
    __shared__ __align__(16) bf16_t Vl[2][DH * 32];    // 2 x  8 KB
    __shared__ __align__(16) bf16_t Pl[QW][16 * 32];   //      8 KB

    const int tid  = threadIdx.x;
    const int lane = tid & 31;
    const int wave = tid >> 5;
    const int h = blockIdx.y, b = blockIdx.z;
    const int q0 = (blockIdx.x * QW + wave) * 16;
    const int qmax_blk = (blockIdx.x * QW + QW - 1) * 16 + 15;  // block max q
    const size_t bh = (size_t)(b * HEADS + h);

    const bf16_t* Qbase = Qf + (bh * S_LEN + q0) * DQK;
    const bf16_t* Kbase = Kf + bh * S_LEN * DQK;
    const bf16_t* Vbase = Vt + bh * (size_t)DH * S_LEN;

    bf16x16 qa[6];
#pragma unroll
    for (int c = 0; c < 6; ++c) qa[c] = load_frag(Qbase + c * 32, DQK, lane);

    f32x8 acc[8];
    float m[8], l[8];
#pragma unroll
    for (int d = 0; d < 8; ++d) acc[d] = zero8();
#pragma unroll
    for (int v = 0; v < 8; ++v) { m[v] = -1e30f; l[v] = 0.0f; }

    const float scale = 0.07216878364870323f;   // 1/sqrt(192)
    const int rowh = (lane >> 4) << 3;
    const int col0 = lane & 15;

    // Stage key tile [k0, k0+32) into buffer `buf`.
    // 32x192 K-tile (768 16B chunks) + 128x32 V-tile (512 chunks) = 5 chunks
    // per thread; K/V split lands on 256-thread boundaries (no divergence).
    auto stage = [&](int k0, int buf) {
        const bf16_t* Kg = Kbase + (size_t)k0 * DQK;
        const bf16_t* Vg = Vbase + k0;
#pragma unroll
        for (int i = 0; i < 5; ++i) {
            int c = tid + i * 256;
            if (c < 768) {
                int row = c / 24, off = (c % 24) * 8;
                cp16(Kg + (size_t)row * DQK + off, &Kl[buf][row * DQK + off]);
            } else {
                int cc = c - 768;
                int d = cc / 4, off = (cc % 4) * 8;
                cp16(Vg + (size_t)d * S_LEN + off, &Vl[buf][d * 32 + off]);
            }
        }
    };

    const int nTiles = qmax_blk / 32 + 1;
    stage(0, 0);
    cp16_drain();
    __syncthreads();

    for (int t = 0; t < nTiles; ++t) {
        const int k0  = t * 32;
        const int buf = t & 1;
        if (t + 1 < nTiles) stage((t + 1) * 32, buf ^ 1);  // async prefetch

        const bool act0 = (k0 <= q0 + 15);
        if (act0) {
            const bool act1 = (k0 + 16 <= q0 + 15);
            float sc[2][8];
#pragma unroll
            for (int sub = 0; sub < 2; ++sub) {
                if (sub == 0 || act1) {
                    f32x8 s = zero8();
#pragma unroll
                    for (int c = 0; c < 6; ++c) {
                        bf16x16 kb = load_frag(&Kl[buf][(sub * 16) * DQK + c * 32],
                                               DQK, lane);
                        s = wmma_bf16(qa[c], kb, s);
                    }
#pragma unroll
                    for (int v = 0; v < 8; ++v) {
                        int qrow = q0 + rowh + v;
                        int kcol = k0 + sub * 16 + col0;
                        float x = fminf(fmaxf(s[v], -80.0f), 80.0f) * scale;
                        sc[sub][v] = (kcol <= qrow) ? x : -1e30f;
                    }
                } else {
#pragma unroll
                    for (int v = 0; v < 8; ++v) sc[sub][v] = -1e30f;
                }
            }
            // online softmax: rows live in 16-lane halves -> xor butterfly
            float rmax[8], rsum[8], alpha[8];
#pragma unroll
            for (int v = 0; v < 8; ++v) rmax[v] = fmaxf(sc[0][v], sc[1][v]);
#pragma unroll
            for (int msk = 1; msk < 16; msk <<= 1)
#pragma unroll
                for (int v = 0; v < 8; ++v)
                    rmax[v] = fmaxf(rmax[v], __shfl_xor(rmax[v], msk, 32));
#pragma unroll
            for (int v = 0; v < 8; ++v) {
                float mn = fmaxf(m[v], rmax[v]);
                alpha[v] = __expf(m[v] - mn);
                m[v] = mn;
                float p0 = __expf(sc[0][v] - mn);
                float p1 = __expf(sc[1][v] - mn);
                int prow = rowh + v;
                Pl[wave][prow * 32 + col0]      = f2bf(p0);
                Pl[wave][prow * 32 + 16 + col0] = f2bf(p1);
                rsum[v] = p0 + p1;
            }
#pragma unroll
            for (int msk = 1; msk < 16; msk <<= 1)
#pragma unroll
                for (int v = 0; v < 8; ++v)
                    rsum[v] += __shfl_xor(rsum[v], msk, 32);
#pragma unroll
            for (int v = 0; v < 8; ++v) l[v] = l[v] * alpha[v] + rsum[v];
#pragma unroll
            for (int d = 0; d < 8; ++d)
#pragma unroll
                for (int v = 0; v < 8; ++v) acc[d][v] *= alpha[v];

            // P @ V from LDS (P relayout C->A through per-wave LDS tile)
            bf16x16 pa = load_frag(Pl[wave], 32, lane);
#pragma unroll
            for (int d = 0; d < 8; ++d) {
                bf16x16 vb = load_frag(&Vl[buf][(d * 16) * 32], 32, lane);
                acc[d] = wmma_bf16(pa, vb, acc[d]);
            }
        }
        cp16_drain();        // own async copies for tile t+1 done
        __syncthreads();     // all waves done reading buf + staging buf^1
    }

    // finalize: attn -> (B,S,H*DH) bf16 for the W_o GEMM
    float inv[8];
#pragma unroll
    for (int v = 0; v < 8; ++v) inv[v] = 1.0f / l[v];
#pragma unroll
    for (int d = 0; d < 8; ++d)
#pragma unroll
        for (int v = 0; v < 8; ++v) {
            size_t tok = (size_t)b * S_LEN + q0 + rowh + v;
            size_t col = (size_t)h * DH + d * 16 + col0;
            Att[tok * (HEADS * DH) + col] = f2bf(acc[d][v] * inv[v]);
        }
}

// ---------------- orchestration ----------------
extern "C" void kernel_launch(void* const* d_in, const int* in_sizes, int n_in,
                              void* d_out, int out_size, void* d_ws, size_t ws_size,
                              hipStream_t stream) {
    (void)in_sizes; (void)n_in; (void)out_size; (void)ws_size;
    const float* x    = (const float*)d_in[0];
    const float* Wdkv = (const float*)d_in[1];
    const float* Wuk  = (const float*)d_in[2];
    const float* Wuv  = (const float*)d_in[3];
    const float* Wdq  = (const float*)d_in[4];
    const float* Wuq  = (const float*)d_in[5];
    const float* Wrq  = (const float*)d_in[6];
    const float* Wrk  = (const float*)d_in[7];
    const float* Wo   = (const float*)d_in[8];
    float* out = (float*)d_out;

    char* ws = (char*)d_ws;
    size_t off = 0;
    auto carve = [&](size_t elems) -> bf16_t* {
        bf16_t* p = (bf16_t*)(ws + off);
        off += (elems * sizeof(bf16_t) + 255) & ~(size_t)255;
        return p;
    };
    bf16_t* Xb    = carve((size_t)NTOK * DMODEL);
    bf16_t* Wdkvb = carve((size_t)DLAT * DMODEL);
    bf16_t* Wukb  = carve((size_t)DMODEL * DLAT);
    bf16_t* Wuvb  = carve((size_t)DMODEL * DLAT);
    bf16_t* Wdqb  = carve((size_t)DLAT * DMODEL);
    bf16_t* Wuqb  = carve((size_t)DMODEL * DLAT);
    bf16_t* Wrqb  = carve((size_t)(HEADS * DHR) * DLAT);
    bf16_t* Wrkb  = carve((size_t)DHR * DMODEL);
    bf16_t* Wob   = carve((size_t)DMODEL * DMODEL);
    bf16_t* cQ    = carve((size_t)NTOK * DLAT);
    bf16_t* cKV   = carve((size_t)NTOK * DLAT);
    bf16_t* Qn    = carve((size_t)NTOK * DMODEL);
    bf16_t* Qr    = carve((size_t)NTOK * HEADS * DHR);
    bf16_t* Kn    = carve((size_t)NTOK * DMODEL);
    bf16_t* Vn    = carve((size_t)NTOK * DMODEL);
    bf16_t* Kr    = carve((size_t)NTOK * DHR);
    bf16_t* Qf    = carve((size_t)B_SZ * HEADS * S_LEN * DQK);
    bf16_t* Kf    = carve((size_t)B_SZ * HEADS * S_LEN * DQK);
    bf16_t* Vt    = carve((size_t)B_SZ * HEADS * DH * S_LEN);
    bf16_t* Att   = carve((size_t)NTOK * HEADS * DH);   // ~190 MB total

    auto cvt = [&](const float* src, bf16_t* dst, long n) {
        int blocks = (int)((n + 255) / 256);
        if (blocks > 4096) blocks = 4096;
        cvt_f32_bf16<<<blocks, 256, 0, stream>>>(src, dst, n);
    };
    cvt(x,    Xb,    (long)NTOK * DMODEL);
    cvt(Wdkv, Wdkvb, (long)DLAT * DMODEL);
    cvt(Wuk,  Wukb,  (long)DMODEL * DLAT);
    cvt(Wuv,  Wuvb,  (long)DMODEL * DLAT);
    cvt(Wdq,  Wdqb,  (long)DLAT * DMODEL);
    cvt(Wuq,  Wuqb,  (long)DMODEL * DLAT);
    cvt(Wrq,  Wrqb,  (long)(HEADS * DHR) * DLAT);
    cvt(Wrk,  Wrkb,  (long)DHR * DMODEL);
    cvt(Wo,   Wob,   (long)DMODEL * DMODEL);

    auto gemm = [&](const bf16_t* A, const bf16_t* W, bf16_t* Cb, float* Cf,
                    int M, int N, int K) {
        dim3 g(N / 64, M / 32);
        gemm_bf16_wmma<<<g, 32, 0, stream>>>(A, W, Cb, Cf, M, N, K);
    };
    // latent projections
    gemm(Xb,  Wdqb,  cQ,  nullptr, NTOK, DLAT,         DMODEL);
    gemm(Xb,  Wdkvb, cKV, nullptr, NTOK, DLAT,         DMODEL);
    // up projections
    gemm(cQ,  Wuqb,  Qn,  nullptr, NTOK, DMODEL,       DLAT);
    gemm(cQ,  Wrqb,  Qr,  nullptr, NTOK, HEADS * DHR,  DLAT);
    gemm(cKV, Wukb,  Kn,  nullptr, NTOK, DMODEL,       DLAT);
    gemm(cKV, Wuvb,  Vn,  nullptr, NTOK, DMODEL,       DLAT);
    gemm(Xb,  Wrkb,  Kr,  nullptr, NTOK, DHR,          DMODEL);

    // RoPE + concat + V transpose
    mla_assemble<<<(B_SZ * HEADS * S_LEN) / 256, 256, 0, stream>>>(
        Qn, Qr, Kn, Kr, Vn, Qf, Kf, Vt);

    // attention: 16 blocks of 8 q-tiles per (b,h)
    dim3 ga(S_LEN / (16 * QW), HEADS, B_SZ);
    mla_attention<<<ga, 256, 0, stream>>>(Qf, Kf, Vt, Att);

    // output projection (fp32 out)
    gemm(Att, Wob, nullptr, out, NTOK, DMODEL, HEADS * DH);
}
